// LLaMATransformerBlock_59279138619559
// MI455X (gfx1250) — compile-verified, matrix operand order checked by reading, not run on hardware
//
#include <hip/hip_runtime.h>
#include <hip/hip_bf16.h>

// ---------------- types ----------------
typedef __bf16 bf16_t;
typedef bf16_t v16bf __attribute__((ext_vector_type(16)));
typedef bf16_t v8bf  __attribute__((ext_vector_type(8)));
typedef float  v8f   __attribute__((ext_vector_type(8)));

#define HID   2048
#define NH    16
#define HD    128
#define INTER 5632
#define SEQ   4096

// Per-lane WMMA 16-bit A/B fragment: lane row = lane&15, K half kb = (lane>>4)*8.
// Elements are two contiguous 8x bf16 runs: [kb..kb+7] and [kb+16..kb+23] (16B each).
__device__ __forceinline__ v16bf frag_ld(const bf16_t* p, int kb) {
    union { v16bf v; v8bf h[2]; } u;
    u.h[0] = *(const v8bf*)(p + kb);
    u.h[1] = *(const v8bf*)(p + kb + 16);
    return u.v;
}

__device__ __forceinline__ v8f wmma_bf16(v16bf a, v16bf b, v8f c) {
    return __builtin_amdgcn_wmma_f32_16x16x32_bf16(false, a, false, b, (short)0, c,
                                                   false, false);
}

// ---------------- elementwise kernels ----------------
// f32 (K,N) -> bf16 transposed (N,K), 32x32 LDS-tiled
__global__ __launch_bounds__(256) void cast_transpose_kernel(const float* __restrict__ in,
                                                             bf16_t* __restrict__ out,
                                                             int K, int N) {
    __shared__ bf16_t tile[32][33];
    const int k0 = blockIdx.y * 32, n0 = blockIdx.x * 32;
    const int tx = threadIdx.x & 31, ty = threadIdx.x >> 5;   // 32x8
#pragma unroll
    for (int r = ty; r < 32; r += 8)
        tile[r][tx] = (bf16_t)in[(size_t)(k0 + r) * N + n0 + tx];
    __syncthreads();
#pragma unroll
    for (int r = ty; r < 32; r += 8)
        out[(size_t)(n0 + r) * K + k0 + tx] = tile[tx][r];
}

__global__ __launch_bounds__(256) void silu_mul_kernel(const float* __restrict__ g,
                                                       const float* __restrict__ u,
                                                       bf16_t* __restrict__ o, int n) {
    int i = blockIdx.x * 256 + threadIdx.x;
    if (i < n) {
        float gv = g[i];
        float a = gv / (1.0f + __expf(-gv));
        o[i] = (bf16_t)(a * u[i]);
    }
}

// One block per row: RMSNorm (f32 in) -> bf16 out
__global__ __launch_bounds__(256) void rmsnorm_cast_kernel(const float* __restrict__ x,
                                                           const float* __restrict__ w,
                                                           bf16_t* __restrict__ o) {
    __shared__ float red[256];
    const int row = blockIdx.x, tid = threadIdx.x;
    const float* xr = x + (size_t)row * HID;
    float ss = 0.f;
    for (int i = tid; i < HID; i += 256) { float v = xr[i]; ss += v * v; }
    red[tid] = ss;
    __syncthreads();
    for (int s = 128; s > 0; s >>= 1) {
        if (tid < s) red[tid] += red[tid + s];
        __syncthreads();
    }
    float inv = rsqrtf(red[0] / (float)HID + 1e-6f);
    bf16_t* orow = o + (size_t)row * HID;
    for (int i = tid; i < HID; i += 256) orow[i] = (bf16_t)(xr[i] * inv * w[i]);
}

// RoPE on f32 (S, NH, HD) interleaved pairs -> bf16
__global__ __launch_bounds__(256) void rope_cast_kernel(const float* __restrict__ q,
                                                        const float* __restrict__ c,
                                                        const float* __restrict__ s,
                                                        bf16_t* __restrict__ o) {
    int idx = blockIdx.x * 256 + threadIdx.x;        // SEQ*NH*64
    int pr = idx & 63;
    int t  = idx >> 6;
    int hh = t & (NH - 1);
    int sq = t >> 4;
    float cv = c[sq * 64 + pr], sv = s[sq * 64 + pr];
    size_t base = (size_t)sq * HID + hh * HD + pr * 2;
    float qr = q[base], qi = q[base + 1];
    o[base]     = (bf16_t)(qr * cv - qi * sv);
    o[base + 1] = (bf16_t)(qr * sv + qi * cv);
}

// ---------------- WMMA GEMM: C(MxN) = A(MxK) * Bt(NxK)^T, bf16 in, f32 accum ----------
// Both operands row-major over K -> identical tile geometry, all-b128 LDS traffic.
// Double-buffered LDS with register staging: one barrier per K-step.
// MODE 0: f32 out; MODE 1: f32 out + f32 residual; MODE 3: bf16 out transposed (N,M)
template <int MODE>
__global__ __launch_bounds__(256) void gemm_wmma_kernel(const bf16_t* __restrict__ A,
                                                        const bf16_t* __restrict__ Bt,
                                                        float* __restrict__ Cf,
                                                        bf16_t* __restrict__ Cb,
                                                        const float* __restrict__ R,
                                                        int M, int N, int K) {
    __shared__ alignas(16) bf16_t As[2][128][48];
    __shared__ alignas(16) bf16_t Bs[2][128][48];
    const int tid  = threadIdx.x;
    const int m0   = blockIdx.y * 128, n0 = blockIdx.x * 128;
    const int wave = tid >> 5, lane = tid & 31;
    const int wm   = wave >> 1, wn = wave & 1;       // 4x2 wave grid, 32x64 per wave
    const int r16  = lane & 15, kb = (lane >> 4) * 8;
    const int lrow = tid >> 2, lkc = tid & 3;        // tile loading: 2 rows/thread

    const v8f vzero = {0.f, 0.f, 0.f, 0.f, 0.f, 0.f, 0.f, 0.f};
    v8f acc[2][4];
    for (int i = 0; i < 2; ++i)
        for (int j = 0; j < 4; ++j) acc[i][j] = vzero;

    const int nkt = K >> 5;
    v8bf a0, a1, b0, b1;

    // preload tile 0 straight into buffer 0
    a0 = *(const v8bf*)(A  + (size_t)(m0 + lrow) * K + lkc * 8);
    a1 = *(const v8bf*)(A  + (size_t)(m0 + lrow + 64) * K + lkc * 8);
    b0 = *(const v8bf*)(Bt + (size_t)(n0 + lrow) * K + lkc * 8);
    b1 = *(const v8bf*)(Bt + (size_t)(n0 + lrow + 64) * K + lkc * 8);
    *(v8bf*)(&As[0][lrow][lkc * 8])      = a0;
    *(v8bf*)(&As[0][lrow + 64][lkc * 8]) = a1;
    *(v8bf*)(&Bs[0][lrow][lkc * 8])      = b0;
    *(v8bf*)(&Bs[0][lrow + 64][lkc * 8]) = b1;
    __syncthreads();

    for (int kt = 0; kt < nkt; ++kt) {
        const int cur = kt & 1, nxt = cur ^ 1;
        const bool more = (kt + 1) < nkt;
        if (more) {                               // stage next tile in registers
            const int k0 = (kt + 1) * 32;
            a0 = *(const v8bf*)(A  + (size_t)(m0 + lrow) * K + k0 + lkc * 8);
            a1 = *(const v8bf*)(A  + (size_t)(m0 + lrow + 64) * K + k0 + lkc * 8);
            b0 = *(const v8bf*)(Bt + (size_t)(n0 + lrow) * K + k0 + lkc * 8);
            b1 = *(const v8bf*)(Bt + (size_t)(n0 + lrow + 64) * K + k0 + lkc * 8);
            if (kt + 2 < nkt) {                   // gfx1250 global_prefetch_b8
                __builtin_prefetch(A  + (size_t)(m0 + lrow) * K + k0 + 32 + lkc * 8, 0, 1);
                __builtin_prefetch(Bt + (size_t)(n0 + lrow) * K + k0 + 32 + lkc * 8, 0, 1);
            }
        }

        v16bf af[2], bfr[4];
#pragma unroll
        for (int mt = 0; mt < 2; ++mt)
            af[mt] = frag_ld(&As[cur][wm * 32 + mt * 16 + r16][0], kb);
#pragma unroll
        for (int nt = 0; nt < 4; ++nt)
            bfr[nt] = frag_ld(&Bs[cur][wn * 64 + nt * 16 + r16][0], kb);
#pragma unroll
        for (int mt = 0; mt < 2; ++mt)
#pragma unroll
            for (int nt = 0; nt < 4; ++nt)
                acc[mt][nt] = wmma_bf16(af[mt], bfr[nt], acc[mt][nt]);

        if (more) {                               // commit staged tile to other buffer
            *(v8bf*)(&As[nxt][lrow][lkc * 8])      = a0;
            *(v8bf*)(&As[nxt][lrow + 64][lkc * 8]) = a1;
            *(v8bf*)(&Bs[nxt][lrow][lkc * 8])      = b0;
            *(v8bf*)(&Bs[nxt][lrow + 64][lkc * 8]) = b1;
        }
        __syncthreads();
    }

    const int hi8 = (lane >> 4) * 8;              // C layout: M = r + 8*(lane>=16), N = lane&15
#pragma unroll
    for (int mt = 0; mt < 2; ++mt)
        for (int nt = 0; nt < 4; ++nt)
#pragma unroll
            for (int r = 0; r < 8; ++r) {
                int grow = m0 + wm * 32 + mt * 16 + r + hi8;
                int gcol = n0 + wn * 64 + nt * 16 + r16;
                float v = acc[mt][nt][r];
                if (MODE == 3) {
                    Cb[(size_t)gcol * M + grow] = (bf16_t)v;   // transposed bf16 out
                } else {
                    size_t idx = (size_t)grow * N + gcol;
                    if (MODE == 1) v += R[idx];
                    Cf[idx] = v;
                }
            }
}

// ---------------- flash attention (causal), bf16 Q/K (S,HID), bf16 Vt (HID,S) ---------
// grid (SEQ/64, NH), 128 threads = 4 waves, each wave owns 16 q rows x full HD.
__global__ __launch_bounds__(128) void flash_attn_kernel(const bf16_t* __restrict__ Q,
                                                         const bf16_t* __restrict__ Kg,
                                                         const bf16_t* __restrict__ Vtg,
                                                         bf16_t* __restrict__ O) {
    __shared__ alignas(16) bf16_t Ks[64][144];    // keys row-major (== B^T layout for QK^T)
    __shared__ alignas(16) bf16_t Vs[128][72];    // Vt tile: Vs[hd][key]
    __shared__ float  st[4][16][68];              // per-wave f32 score tile 16x64
    __shared__ alignas(16) bf16_t pt[4][16][80];  // per-wave bf16 P tile 16x64
    __shared__ float rowm[4][16], rowl[4][16], rsc[4][16];

    const int qb0  = blockIdx.x * 64;
    const int h    = blockIdx.y;
    const int tid  = threadIdx.x;
    const int wave = tid >> 5, lane = tid & 31;
    const int r16  = lane & 15, kb = (lane >> 4) * 8, hi8 = (lane >> 4) * 8;
    const float scale = 0.08838834764831845f;     // 1/sqrt(HD)

    const bf16_t* qrow = Q + (size_t)(qb0 + wave * 16 + r16) * HID + h * HD;
    v16bf qf[4];
#pragma unroll
    for (int kc = 0; kc < 4; ++kc) qf[kc] = frag_ld(qrow + kc * 32, kb);

    const v8f vzero = {0.f, 0.f, 0.f, 0.f, 0.f, 0.f, 0.f, 0.f};
    v8f acc[8];
#pragma unroll
    for (int i = 0; i < 8; ++i) acc[i] = vzero;
    if (lane < 16) { rowm[wave][r16] = -3.0e38f; rowl[wave][r16] = 0.f; }

    const int jend = blockIdx.x;
    for (int j = 0; j <= jend; ++j) {
        // cooperative loads: K rows (64x128) and Vt rows (128x64), all b128
#pragma unroll
        for (int t = 0; t < 8; ++t) {
            int idx = tid + t * 128;
            int row = idx >> 4, nc = idx & 15;
            v8bf kv = *(const v8bf*)(Kg + (size_t)(j * 64 + row) * HID + h * HD + nc * 8);
            *(v8bf*)(&Ks[row][nc * 8]) = kv;
        }
#pragma unroll
        for (int t = 0; t < 8; ++t) {
            int idx = tid + t * 128;
            int d = idx >> 3, c = idx & 7;
            v8bf vv = *(const v8bf*)(Vtg + (size_t)(h * HD + d) * SEQ + j * 64 + c * 8);
            *(v8bf*)(&Vs[d][c * 8]) = vv;
        }
        __syncthreads();

        // S = Q K^T : per wave 16x64 (4 n-subtiles x 4 k-chunks)
#pragma unroll
        for (int nt = 0; nt < 4; ++nt) {
            v8f s = vzero;
#pragma unroll
            for (int kc = 0; kc < 4; ++kc) {
                v16bf kf = frag_ld(&Ks[nt * 16 + r16][kc * 32], kb);
                s = wmma_bf16(qf[kc], kf, s);
            }
#pragma unroll
            for (int r = 0; r < 8; ++r)
                st[wave][r + hi8][nt * 16 + r16] = s[r] * scale;
        }
        __syncthreads();

        // online softmax, one lane per q row
        if (lane < 16) {
            int grow = qb0 + wave * 16 + r16;
            int kmax = grow - j * 64 + 1;
            if (kmax > 64) kmax = 64;
            float mold = rowm[wave][r16];
            float mnew = mold;
            for (int c = 0; c < kmax; ++c) mnew = fmaxf(mnew, st[wave][r16][c]);
            float sc = __expf(mold - mnew);
            float ls = 0.f;
            for (int c = 0; c < 64; ++c) {
                float p = (c < kmax) ? __expf(st[wave][r16][c] - mnew) : 0.f;
                pt[wave][r16][c] = (bf16_t)p;
                ls += p;
            }
            rowl[wave][r16] = rowl[wave][r16] * sc + ls;
            rowm[wave][r16] = mnew;
            rsc[wave][r16]  = sc;
        }
        __syncthreads();

        // rescale accumulators, then O += P * V
#pragma unroll
        for (int nt = 0; nt < 8; ++nt)
#pragma unroll
            for (int r = 0; r < 8; ++r) acc[nt][r] *= rsc[wave][r + hi8];
#pragma unroll
        for (int kc2 = 0; kc2 < 2; ++kc2) {
            v16bf pa = frag_ld(&pt[wave][r16][kc2 * 32], kb);
#pragma unroll
            for (int nt = 0; nt < 8; ++nt) {
                v16bf vf = frag_ld(&Vs[nt * 16 + r16][kc2 * 32], kb);
                acc[nt] = wmma_bf16(pa, vf, acc[nt]);
            }
        }
        __syncthreads();
    }

    // normalize and write out (S, NH*HD) == (S, HID)
#pragma unroll
    for (int nt = 0; nt < 8; ++nt)
        for (int r = 0; r < 8; ++r) {
            int m = r + hi8;
            float o = acc[nt][r] / rowl[wave][m];
            O[(size_t)(qb0 + wave * 16 + m) * HID + h * HD + nt * 16 + r16] = (bf16_t)o;
        }
}

// ---------------- host orchestration ----------------
extern "C" void kernel_launch(void* const* d_in, const int* in_sizes, int n_in,
                              void* d_out, int out_size, void* d_ws, size_t ws_size,
                              hipStream_t stream) {
    (void)in_sizes; (void)n_in; (void)out_size; (void)ws_size;
    const float* x  = (const float*)d_in[0];
    const float* fc = (const float*)d_in[1];
    const float* fs = (const float*)d_in[2];
    /* d_in[3] = mask: causal handled analytically */
    const float* wq = (const float*)d_in[4];
    const float* wk = (const float*)d_in[5];
    const float* wv = (const float*)d_in[6];
    const float* wo = (const float*)d_in[7];
    const float* wg = (const float*)d_in[8];
    const float* wu = (const float*)d_in[9];
    const float* wd = (const float*)d_in[10];
    const float* n1 = (const float*)d_in[11];
    const float* n2 = (const float*)d_in[12];

    const size_t HH = (size_t)HID * HID;      // 4194304
    const size_t HI = (size_t)HID * INTER;    // 11534336
    const size_t SH = (size_t)SEQ * HID;      // 8388608
    const size_t SI = (size_t)SEQ * INTER;    // 23068672

    char* ws = (char*)d_ws;
    size_t cur = 0;
    auto alloc = [&](size_t bytes) -> char* {
        char* p = ws + cur;
        cur += (bytes + 255) & ~(size_t)255;
        return p;
    };
    // all weights stored TRANSPOSED (N x K) in bf16
    bf16_t* wqT = (bf16_t*)alloc(HH * 2);
    bf16_t* wkT = (bf16_t*)alloc(HH * 2);
    bf16_t* wvT = (bf16_t*)alloc(HH * 2);
    bf16_t* woT = (bf16_t*)alloc(HH * 2);
    bf16_t* wgT = (bf16_t*)alloc(HI * 2);
    bf16_t* wuT = (bf16_t*)alloc(HI * 2);
    bf16_t* wdT = (bf16_t*)alloc(HI * 2);
    bf16_t* xnb = (bf16_t*)alloc(SH * 2);
    float*  qf  = (float*) alloc(SH * 4);
    float*  kf  = (float*) alloc(SH * 4);
    bf16_t* qb  = (bf16_t*)alloc(SH * 2);
    bf16_t* kbb = (bf16_t*)alloc(SH * 2);
    bf16_t* vtb = (bf16_t*)alloc(SH * 2);     // V^T: (HID, SEQ)
    bf16_t* atb = (bf16_t*)alloc(SH * 2);
    float*  hbf = (float*) alloc(SH * 4);
    bf16_t* hnb = (bf16_t*)alloc(SH * 2);
    float*  gte = (float*) alloc(SI * 4);
    float*  upb = (float*) alloc(SI * 4);
    bf16_t* act = (bf16_t*)alloc(SI * 2);

    // weights -> bf16, transposed to (N, K)
    dim3 tHH(HID / 32, HID / 32);
    cast_transpose_kernel<<<tHH, 256, 0, stream>>>(wq, wqT, HID, HID);
    cast_transpose_kernel<<<tHH, 256, 0, stream>>>(wk, wkT, HID, HID);
    cast_transpose_kernel<<<tHH, 256, 0, stream>>>(wv, wvT, HID, HID);
    cast_transpose_kernel<<<tHH, 256, 0, stream>>>(wo, woT, HID, HID);
    dim3 tHI(INTER / 32, HID / 32);           // (HID,INTER) -> (INTER,HID)
    cast_transpose_kernel<<<tHI, 256, 0, stream>>>(wg, wgT, HID, INTER);
    cast_transpose_kernel<<<tHI, 256, 0, stream>>>(wu, wuT, HID, INTER);
    dim3 tIH(HID / 32, INTER / 32);           // (INTER,HID) -> (HID,INTER)
    cast_transpose_kernel<<<tIH, 256, 0, stream>>>(wd, wdT, INTER, HID);

    // norm1
    rmsnorm_cast_kernel<<<SEQ, 256, 0, stream>>>(x, n1, xnb);

    // QKV projections (V written transposed for flash)
    dim3 gHID(HID / 128, SEQ / 128);
    gemm_wmma_kernel<0><<<gHID, 256, 0, stream>>>(xnb, wqT, qf, nullptr, nullptr, SEQ, HID, HID);
    gemm_wmma_kernel<0><<<gHID, 256, 0, stream>>>(xnb, wkT, kf, nullptr, nullptr, SEQ, HID, HID);
    gemm_wmma_kernel<3><<<gHID, 256, 0, stream>>>(xnb, wvT, nullptr, vtb, nullptr, SEQ, HID, HID);

    // RoPE + cast
    rope_cast_kernel<<<(SEQ * NH * 64) / 256, 256, 0, stream>>>(qf, fc, fs, qb);
    rope_cast_kernel<<<(SEQ * NH * 64) / 256, 256, 0, stream>>>(kf, fc, fs, kbb);

    // causal flash attention
    flash_attn_kernel<<<dim3(SEQ / 64, NH), 128, 0, stream>>>(qb, kbb, vtb, atb);

    // h = x + attn @ wo (residual fused in epilogue)
    gemm_wmma_kernel<1><<<gHID, 256, 0, stream>>>(atb, woT, hbf, nullptr, x, SEQ, HID, HID);

    // norm2
    rmsnorm_cast_kernel<<<SEQ, 256, 0, stream>>>(hbf, n2, hnb);

    // FFN
    dim3 gINT(INTER / 128, SEQ / 128);
    gemm_wmma_kernel<0><<<gINT, 256, 0, stream>>>(hnb, wgT, gte, nullptr, nullptr, SEQ, INTER, HID);
    gemm_wmma_kernel<0><<<gINT, 256, 0, stream>>>(hnb, wuT, upb, nullptr, nullptr, SEQ, INTER, HID);
    silu_mul_kernel<<<SI / 256, 256, 0, stream>>>(gte, upb, act, (int)SI);

    // out = h + act @ w_down (residual fused)
    gemm_wmma_kernel<1><<<gHID, 256, 0, stream>>>(act, wdT, (float*)d_out, nullptr, hbf,
                                                  SEQ, HID, INTER);
}
